// VectorQuantisation_56083682951239
// MI455X (gfx1250) — matrix operand order repaired and checked by reading, not compiled, not run on hardware
//
#include <hip/hip_runtime.h>

// ---------------------------------------------------------------------------
// VQ-VAE vector quantisation for MI455X (gfx1250, wave32, WMMA).
//
//   z_e : [64, 256, 32, 32] fp32  -> flat [N=65536, K=256]
//   w   : [1024, 256] fp32 codebook
// Core: argmin_j (|w_j|^2 - 2 * flat . w_j)  == 65536x1024x256 GEMM + argmin,
// bf16 inputs / fp32 accum via v_wmma_f32_16x16x32_bf16.
//  - codebook streamed through LDS with CDNA5 async global->LDS DMA
//    (ASYNCcnt), double buffered: HBM/L2 latency overlaps 32 WMMAs.
//  - B fragments fetched through a 3-slot register ring (load frag s+2 while
//    WMMA consumes frag s): no copies, so the allocator cannot re-serialize
//    the pipeline; each LDS round-trip gets ~2 WMMAs of cover.
//  - __launch_bounds__(256,1): ~140 VGPRs/lane live (A register-resident)
//    without scratch spills; latency hiding from 8 waves/WG.
//  - in-loop argmin uses strict '<' only (per-lane candidate codes increase
//    monotonically -> first-min-wins preserved); index tie-break only in the
//    final cross-lane reduction.
// Outputs: z_q (transposed-reshape gather, faithful to reference) then
//          vq_loss = 1.25 * mean((z_q - z_e)^2).
// ---------------------------------------------------------------------------

typedef __attribute__((ext_vector_type(16))) __bf16        v16bf;
typedef __attribute__((ext_vector_type(8)))  float         v8f;
typedef __attribute__((ext_vector_type(4)))  unsigned int  u32x4;
typedef __attribute__((ext_vector_type(8)))  unsigned int  u32x8;
typedef unsigned short u16;
typedef unsigned int   u32;
typedef unsigned long long u64;

#define N_ROWS   65536      // 64*32*32
#define K_DIM    256
#define N_CODES  1024
#define N_ELEMS  16777216u  // 64*256*32*32

#define CHUNK_CODES 64                          // codes staged per LDS buffer
#define N_CHUNKS    (N_CODES / CHUNK_CODES)     // 16

__device__ inline u16 f2bf(float x) {
  u32 u = __builtin_bit_cast(u32, x);
  u32 r = u + 0x7FFFu + ((u >> 16) & 1u);   // round-to-nearest-even
  return (u16)(r >> 16);
}

__device__ inline float wave_sum(float v) {
  #pragma unroll
  for (int m = 16; m >= 1; m >>= 1) v += __shfl_xor(v, m, 32);
  return v;
}

// Async DMA of one 32 KB codebook chunk into LDS. 256 threads x 16 B x 8.
// Global and LDS addresses advance identically, so the shared INST_OFFSET
// (added to both per the ISA pseudocode) covers the strided copies.
__device__ inline void async_copy_chunk(u32 lds_base, const u16* gsrc, int t) {
  u64 g = (u64)gsrc + (u32)t * 16u;
  u32 l = lds_base + (u32)t * 16u;
  #pragma unroll
  for (int i = 0; i < 8; ++i) {
    asm volatile("global_load_async_to_lds_b128 %0, %1, off offset:%2"
                 :: "v"(l), "v"(g), "i"(i * 4096) : "memory");
  }
}

__device__ inline void wait_async0() {
  asm volatile("s_wait_asynccnt 0x0" ::: "memory");
}

// B fragment f (f = tile*8 + ktile) of a chunk: 32x16 bf16 K-tile, lanes 0-15
// hold K=0..15, lanes 16-31 hold K=16..31 -> one 32 B LDS read per lane.
__device__ inline v16bf ldB(const u16* lbuf, int f, int col, int half) {
  const int lt = f >> 3, kt = f & 7;
  const u16* p = lbuf + (size_t)(lt * 16 + col) * K_DIM + kt * 32 + half * 16;
  return __builtin_bit_cast(v16bf, *(const u32x8*)p);
}

// --------------------------- 1. codebook prep ------------------------------
// grid 1024, block 256: bf16 copy of w + squared row norms.
__global__ void vq_prep_codes(const float* __restrict__ w,
                              u16* __restrict__ wbf,
                              float* __restrict__ wnorm) {
  int j = blockIdx.x, t = threadIdx.x;
  float x = w[j * K_DIM + t];
  wbf[j * K_DIM + t] = f2bf(x);
  float s = wave_sum(x * x);
  __shared__ float ls[8];
  if ((t & 31) == 0) ls[t >> 5] = s;
  __syncthreads();
  if (t == 0) {
    float tot = 0.f;
    #pragma unroll
    for (int k = 0; k < 8; ++k) tot += ls[k];
    wnorm[j] = tot;
  }
}

// --------------------- 2. z_e -> flat bf16 (transpose) ---------------------
// grid (32, 8, 64), block (32, 32). flat[n][c] = bf16(z_e[b][c][h][w]),
// n = b*1024 + h*32 + w. Coalesced read & write via LDS tile.
__global__ void vq_stage_flat(const float* __restrict__ z_e,
                              u16* __restrict__ flat) {
  __shared__ u16 tile[32][33];
  int b = blockIdx.z, c0 = blockIdx.y * 32, hw0 = blockIdx.x * 32;
  int tx = threadIdx.x, ty = threadIdx.y;
  float v = z_e[b * 262144 + (c0 + ty) * 1024 + hw0 + tx];
  tile[ty][tx] = f2bf(v);                       // tile[c][hw]
  __syncthreads();
  flat[(size_t)(b * 1024 + hw0 + ty) * K_DIM + c0 + tx] = tile[tx][ty];
}

// ------------------ 3. distance GEMM + argmin (WMMA core) ------------------
// grid 512, block 256 (8 waves). Each wave: 16 rows x all 1024 codes.
// A (16x256 bf16) register-resident; B double-buffered in LDS via async DMA.
__global__ void __launch_bounds__(256, 1)
vq_argmin(const u16* __restrict__ flat,
          const u16* __restrict__ wbf,
          const float* __restrict__ wnorm,
          int* __restrict__ idx) {
  __shared__ u16 bbuf[2][CHUNK_CODES * K_DIM];   // 2 x 32 KB

  const int t    = threadIdx.x;
  const int lane = t & 31;
  const int wv   = t >> 5;
  const int m0   = (blockIdx.x * 8 + wv) * 16;
  const int half = lane >> 4;        // 0: lanes 0-15, 1: lanes 16-31
  const int col  = lane & 15;        // N (code within tile) / A row

  const u32 lds0 = (u32)(u64)(const void*)&bbuf[0][0];
  const u32 lds1 = (u32)(u64)(const void*)&bbuf[1][0];

  // ---- prologue: start DMA of chunk 0 while we load A from global.
  async_copy_chunk(lds0, wbf, t);

  // ---- load A: contiguous K layout matching B: VGPR0-7 hold
  //      K = half*16 .. +15 (per 32-wide K tile).
  const u16* arow = flat + (size_t)(m0 + col) * K_DIM;
  v16bf a[8];
  #pragma unroll
  for (int kt = 0; kt < 8; ++kt) {
    int base = kt * 32 + half * 16;
    u32x8 tv;
    tv.lo = *(const u32x4*)(arow + base);
    tv.hi = *(const u32x4*)(arow + base + 8);
    a[kt] = __builtin_bit_cast(v16bf, tv);
  }

  float bestD[8];
  int   bestI[8];
  #pragma unroll
  for (int v = 0; v < 8; ++v) { bestD[v] = 3.4e38f; bestI[v] = 0; }

  wait_async0();
  __syncthreads();                   // chunk 0 resident in bbuf[0]

  for (int c = 0; c < N_CHUNKS; ++c) {
    const int cur = c & 1;
    // ---- kick off DMA of the next chunk into the other buffer.
    if (c + 1 < N_CHUNKS)
      async_copy_chunk(cur ? lds0 : lds1,
                       wbf + (size_t)(c + 1) * CHUNK_CODES * K_DIM, t);

    // ---- preload this chunk's code norms (consumed after each tile; the
    //      global latency hides under the 32 WMMAs below).
    float wn[4];
    #pragma unroll
    for (int lt = 0; lt < 4; ++lt)
      wn[lt] = wnorm[c * CHUNK_CODES + lt * 16 + col];

    const u16* lbuf = &bbuf[cur][0];

    // ---- 3-slot register ring over the 32 fragments of this chunk:
    //      load frag s+2 into slot (s+2)%3 while WMMA s reads slot s%3.
    //      No copies -> the allocator keeps three distinct live registers,
    //      so every ds_load is issued two WMMAs ahead of its use.
    v16bf b0 = ldB(lbuf, 0, col, half);
    v16bf b1 = ldB(lbuf, 1, col, half);
    v16bf b2;
    #pragma unroll
    for (int lt = 0; lt < 4; ++lt) {           // 4 code tiles per chunk
      v8f acc = {0.f, 0.f, 0.f, 0.f, 0.f, 0.f, 0.f, 0.f};
      #pragma unroll
      for (int kt = 0; kt < 8; ++kt) {
        const int s = lt * 8 + kt;
        const int wslot = (s + 2) % 3;         // slot being refilled
        v16bf ldv = ldB(lbuf, (s + 2) & 31, col, half);  // wraps harmlessly
        if (wslot == 0)      b0 = ldv;
        else if (wslot == 1) b1 = ldv;
        else                 b2 = ldv;
        const int rslot = s % 3;               // slot consumed now
        v16bf cb = (rslot == 0) ? b0 : ((rslot == 1) ? b1 : b2);
        acc = __builtin_amdgcn_wmma_f32_16x16x32_bf16(
            false, a[kt], false, cb, (short)0, acc, false, false);
      }
      const int code = c * CHUNK_CODES + lt * 16 + col;
      #pragma unroll
      for (int v = 0; v < 8; ++v) {
        // dist = |w|^2 - 2*dot  (|flat|^2 dropped: row-constant).
        // Strict '<' keeps the earliest code on ties: per-lane candidate
        // codes increase monotonically across the whole loop.
        float d = __builtin_fmaf(-2.f, acc[v], wn[lt]);
        if (d < bestD[v]) { bestD[v] = d; bestI[v] = code; }
      }
    }
    // ---- our async issues are done; barrier makes next buffer visible and
    //      guarantees everyone finished reading the buffer we overwrite next.
    wait_async0();
    __syncthreads();
  }

  // ---- reduce over the 16 lanes of each half (columns 0..15 of each tile).
  // C layout: acc[v] = row (m0 + half*8 + v), col (lane&15). Index tie-break
  // here matches jnp.argmin's first-occurrence rule across lanes.
  #pragma unroll
  for (int m = 8; m >= 1; m >>= 1) {
    #pragma unroll
    for (int v = 0; v < 8; ++v) {
      float od = __shfl_xor(bestD[v], m, 32);
      int   oi = __shfl_xor(bestI[v], m, 32);
      if (od < bestD[v] || (od == bestD[v] && oi < bestI[v])) {
        bestD[v] = od; bestI[v] = oi;
      }
    }
  }
  if (col == 0) {
    int rbase = m0 + half * 8;
    #pragma unroll
    for (int v = 0; v < 8; ++v) idx[rbase + v] = bestI[v];
  }
}

// ------------------- 4. gather (transposed reshape) + loss -----------------
// out.flat[p] = w[idx[p & 65535], p >> 16]; per-block deterministic partial
// sums of (z_q - z_e)^2. grid 4096, block 256, 16 elems/thread; `i` is
// constant per block since 4096 | 65536.
__global__ void vq_gather_loss(const float* __restrict__ w,
                               const int* __restrict__ idx,
                               const float* __restrict__ z_e,
                               float* __restrict__ out,
                               float* __restrict__ partials) {
  const int t = threadIdx.x;
  const u32 base = blockIdx.x * 4096u;
  const u32 i = base >> 16;                 // codebook column for this block
  float s = 0.f;
  #pragma unroll
  for (int k = 0; k < 16; ++k) {
    u32 p = base + (u32)k * 256u + t;
    u32 j = p & 65535u;
    float q = w[(u32)idx[j] * 256u + i];    // L2-resident gather (1 MB table)
    float d = q - z_e[p];
    out[p] = q;
    s += d * d;
  }
  s = wave_sum(s);
  __shared__ float ls[8];
  if ((t & 31) == 0) ls[t >> 5] = s;
  __syncthreads();
  if (t == 0) {
    float tot = 0.f;
    #pragma unroll
    for (int k = 0; k < 8; ++k) tot += ls[k];
    partials[blockIdx.x] = tot;
  }
}

// ------------------------- 5. deterministic finish -------------------------
__global__ void vq_finalize(const float* __restrict__ partials,
                            float* __restrict__ loss) {
  const int t = threadIdx.x;
  float s = 0.f;
  #pragma unroll
  for (int k = 0; k < 16; ++k) s += partials[t * 16 + k];
  __shared__ float ls[256];
  ls[t] = s;
  __syncthreads();
  for (int off = 128; off >= 1; off >>= 1) {
    if (t < off) ls[t] += ls[t + off];
    __syncthreads();
  }
  // dict loss + 0.25 * commitment loss; forward diffs are identical.
  if (t == 0) *loss = ls[0] * (1.25f / (float)N_ELEMS);
}

// ---------------------------------------------------------------------------
extern "C" void kernel_launch(void* const* d_in, const int* in_sizes, int n_in,
                              void* d_out, int out_size, void* d_ws, size_t ws_size,
                              hipStream_t stream) {
  const float* z_e = (const float*)d_in[0];   // 16,777,216 fp32
  const float* w   = (const float*)d_in[1];   //    262,144 fp32
  float* out = (float*)d_out;                 // z_q (16,777,216) + loss (1)

  // Scratch layout: flat bf16 + codebook staging live in d_out (fully
  // overwritten by the gather pass afterwards); only idx/partials need d_ws.
  char* ob = (char*)d_out;
  u16*   flat  = (u16*)ob;                                   // 32 MB
  u16*   wbf   = (u16*)(ob + (32u << 20));                   // 512 KB
  float* wnorm = (float*)(ob + (32u << 20) + (512u << 10));  // 4 KB
  int*   idx      = (int*)d_ws;                              // 256 KB
  float* partials = (float*)((char*)d_ws + (256u << 10));    // 16 KB

  vq_prep_codes<<<N_CODES, 256, 0, stream>>>(w, wbf, wnorm);
  vq_stage_flat<<<dim3(32, 8, 64), dim3(32, 32), 0, stream>>>(z_e, flat);
  vq_argmin<<<N_ROWS / 128, 256, 0, stream>>>(flat, wbf, wnorm, idx);
  vq_gather_loss<<<N_ELEMS / 4096u, 256, 0, stream>>>(w, idx, z_e, out, partials);
  vq_finalize<<<1, 256, 0, stream>>>(partials, out + N_ELEMS);
}
